// TFGPTNeoSelfAttention_50345606643994
// MI455X (gfx1250) — matrix operand order repaired and checked by reading, not compiled, not run on hardware
//
#include <hip/hip_runtime.h>

typedef __attribute__((ext_vector_type(4)))  __bf16 bf16x4;
typedef __attribute__((ext_vector_type(8)))  __bf16 bf16x8;
typedef __attribute__((ext_vector_type(16))) __bf16 bf16x16;
typedef __attribute__((ext_vector_type(8)))  float  f32x8;
typedef int v4i __attribute__((vector_size(16)));  // matches async-LDS builtin

union Frag { bf16x16 v; bf16x8 h[2]; };

__device__ inline f32x8 wmma_bf16(bf16x16 a, bf16x16 b, f32x8 c) {
  // D = A(16x32 bf16) * B(32x16 bf16) + C(16x16 f32)
  return __builtin_amdgcn_wmma_f32_16x16x32_bf16(
      /*neg_a=*/false, a, /*neg_b=*/false, b,
      /*c_mod=*/(short)0, c, /*reuse_a=*/false, /*reuse_b=*/false);
}

__device__ inline f32x8 zero8() {
  f32x8 z = {0.f, 0.f, 0.f, 0.f, 0.f, 0.f, 0.f, 0.f};
  return z;
}

// ---------------------------------------------------------------------------
// Async global -> LDS 16B copy (CDNA5 ASYNCcnt path), with sync fallback.
// Builtin signature (from diagnostics): (v4i AS1*, v4i AS3*, imm off, imm cpol).
// AS1 ptr = 64-bit VA; AS3 ptr = low-32 LDS offset (ISA generic->LDS rule).
// ---------------------------------------------------------------------------
#if defined(__gfx1250__) && __has_builtin(__builtin_amdgcn_global_load_async_to_lds_b128)
#define HAVE_ASYNC_LDS 1
#else
#define HAVE_ASYNC_LDS 0
#endif

__device__ inline void cp16_async(__bf16* dst_lds, const __bf16* src_g) {
#if HAVE_ASYNC_LDS
  __builtin_amdgcn_global_load_async_to_lds_b128(
      (__attribute__((address_space(1))) v4i*)(unsigned long long)(uintptr_t)src_g,
      (__attribute__((address_space(3))) v4i*)(unsigned)(uintptr_t)dst_lds,
      0, 0);
#else
  *(bf16x8*)dst_lds = *(const bf16x8*)src_g;
#endif
}

__device__ inline void async_wait0() {
#if HAVE_ASYNC_LDS
#if __has_builtin(__builtin_amdgcn_s_wait_asynccnt)
  __builtin_amdgcn_s_wait_asynccnt(0);
#else
  asm volatile("s_wait_asynccnt 0x0" ::: "memory");
#endif
#endif
}

// ---------------------------------------------------------------------------
// f32 -> bf16 bulk convert (vectorized by 4)
// ---------------------------------------------------------------------------
__global__ __launch_bounds__(256) void cvt_f32_bf16(
    const float* __restrict__ x, __bf16* __restrict__ y, int n) {
  int i = (blockIdx.x * 256 + threadIdx.x) * 4;
  if (i + 3 < n) {
    float4 v = *(const float4*)(x + i);
    bf16x4 o;
    o[0] = (__bf16)v.x; o[1] = (__bf16)v.y; o[2] = (__bf16)v.z; o[3] = (__bf16)v.w;
    *(bf16x4*)(y + i) = o;
  }
}

// ---------------------------------------------------------------------------
// Wt[n][k] = (bf16) W[k][n]   (64x64 LDS tile transpose)
// ---------------------------------------------------------------------------
__global__ __launch_bounds__(256) void transpose_cvt(
    const float* __restrict__ W, __bf16* __restrict__ Wt, int K, int N) {
  __shared__ __bf16 tile[64][72];
  const int k0 = blockIdx.y * 64;
  const int n0 = blockIdx.x * 64;
  const int tx = threadIdx.x & 63;
  const int ty = threadIdx.x >> 6;  // 0..3
#pragma unroll
  for (int i = 0; i < 16; ++i) {
    int r = ty + i * 4;
    tile[r][tx] = (__bf16)W[(size_t)(k0 + r) * N + (n0 + tx)];
  }
  __syncthreads();
#pragma unroll
  for (int i = 0; i < 16; ++i) {
    int r = ty + i * 4;  // n-row within tile
    Wt[(size_t)(n0 + r) * K + (k0 + tx)] = tile[tx][r];
  }
}

// ---------------------------------------------------------------------------
// C[M][N] = A[M][K] @ Bt[N][K]^T   (bf16 WMMA, f32 accum)
// Block tile 128x128, 8 waves (2M x 4N), wave tile 64x32, K-step 32,
// double-buffered LDS staged with async global->LDS copies.
// If Cf != null: f32 output with bias; else bf16 output to Cb.
// ---------------------------------------------------------------------------
__global__ __launch_bounds__(256) void gemm_bf16_wmma(
    const __bf16* __restrict__ A,    // [M][K]
    const __bf16* __restrict__ Bt,   // [N][K]  (B transposed)
    __bf16* __restrict__ Cb,         // [M][N] bf16 out (or null)
    float* __restrict__ Cf,          // [M][N] f32 out  (or null)
    const float* __restrict__ bias,  // [N] (used when Cf != null)
    int M, int N, int K) {
  __shared__ __bf16 As[2][128][40];  // row stride 80B (16B aligned), 2 buffers
  __shared__ __bf16 Bs[2][128][40];

  const int tid   = threadIdx.x;
  const int lane  = tid & 31;
  const int wave  = tid >> 5;     // 0..7
  const int wm    = wave >> 2;    // 0..1  -> +wm*64 in M
  const int wn    = wave & 3;     // 0..3  -> +wn*32 in N
  const int l15   = lane & 15;
  const int lhalf = lane >> 4;    // 0/1
  const int m0    = blockIdx.y * 128;
  const int n0    = blockIdx.x * 128;

  f32x8 acc[4][2];
#pragma unroll
  for (int mi = 0; mi < 4; ++mi)
#pragma unroll
    for (int ni = 0; ni < 2; ++ni) acc[mi][ni] = zero8();

  auto stage = [&](int buf, int k0) {
#pragma unroll
    for (int c = tid; c < 512; c += 256) {
      int r = c >> 2, off = (c & 3) * 8;
      cp16_async(&As[buf][r][off], &A [(size_t)(m0 + r) * K + k0 + off]);
      cp16_async(&Bs[buf][r][off], &Bt[(size_t)(n0 + r) * K + k0 + off]);
    }
  };

  const int nk = K >> 5;
  stage(0, 0);
  async_wait0();
  __syncthreads();

  int buf = 0;
  for (int kt = 0; kt < nk; ++kt) {
    if (kt + 1 < nk) stage(buf ^ 1, (kt + 1) * 32);  // prefetch under compute

    Frag bfr[2];
#pragma unroll
    for (int ni = 0; ni < 2; ++ni) {
      const __bf16* bp = &Bs[buf][wn * 32 + ni * 16 + l15][lhalf * 8];
      bfr[ni].h[0] = *(const bf16x8*)bp;
      bfr[ni].h[1] = *(const bf16x8*)(bp + 16);
    }
#pragma unroll
    for (int mi = 0; mi < 4; ++mi) {
      Frag a;
      const __bf16* ap = &As[buf][wm * 64 + mi * 16 + l15][lhalf * 8];
      a.h[0] = *(const bf16x8*)ap;
      a.h[1] = *(const bf16x8*)(ap + 16);
#pragma unroll
      for (int ni = 0; ni < 2; ++ni)
        acc[mi][ni] = wmma_bf16(a.v, bfr[ni].v, acc[mi][ni]);
    }

    async_wait0();   // next-tile LDS writes landed
    __syncthreads(); // everyone done reading `buf`, next buffer ready
    buf ^= 1;
  }

  // Store: C-layout rows = r + lhalf*8, cols = l15.
#pragma unroll
  for (int mi = 0; mi < 4; ++mi)
#pragma unroll
    for (int ni = 0; ni < 2; ++ni) {
      int col = n0 + wn * 32 + ni * 16 + l15;
      int rb  = m0 + wm * 64 + mi * 16 + lhalf * 8;
      if (Cf) {
        float bv = bias ? bias[col] : 0.f;
#pragma unroll
        for (int r = 0; r < 8; ++r)
          Cf[(size_t)(rb + r) * N + col] = acc[mi][ni][r] + bv;
      } else {
#pragma unroll
        for (int r = 0; r < 8; ++r)
          Cb[(size_t)(rb + r) * N + col] = (__bf16)acc[mi][ni][r];
      }
    }
}

// ---------------------------------------------------------------------------
// Causal flash attention. Q/K/V/O: bf16 [B*S][D], head h = cols h*128..+127.
// Grid (S/128, H, B), 256 threads. Wave w owns query rows q0+w*16..+15.
// No 1/sqrt(hd) scaling (GPT-Neo).
// ---------------------------------------------------------------------------
#define HD 128
__global__ __launch_bounds__(256) void attn_flash_wmma(
    const __bf16* __restrict__ Q, const __bf16* __restrict__ Kf,
    const __bf16* __restrict__ Vf, __bf16* __restrict__ O,
    int S, int D, int H) {
  __shared__ __bf16 Ks[128][136];      // [key][hd]   == B-frag layout for Q@K^T
  __shared__ __bf16 Vt[HD][136];       // [hd][key]   == B-frag layout for P@V
  __shared__ __bf16 Ps[8][16][136];    // per-wave P staging for A-frag re-layout

  const int b    = blockIdx.z;
  const int h    = blockIdx.y;
  const int q0   = blockIdx.x * 128;
  const int hoff = h * HD;
  const int tid = threadIdx.x, lane = tid & 31, wave = tid >> 5;
  const int l15 = lane & 15, lhalf = lane >> 4;

  // Q fragments: 16 rows x 128 hd, 4 K-chunks of 32, straight from global.
  Frag qf[4];
  {
    size_t qrow = (size_t)(b * S + q0 + wave * 16 + l15) * D + hoff;
#pragma unroll
    for (int kc = 0; kc < 4; ++kc) {
      const __bf16* qp = &Q[qrow + kc * 32 + lhalf * 8];
      qf[kc].h[0] = *(const bf16x8*)qp;
      qf[kc].h[1] = *(const bf16x8*)(qp + 16);
    }
  }

  f32x8 ctx[8];
#pragma unroll
  for (int nh = 0; nh < 8; ++nh) ctx[nh] = zero8();
  float mrow[8], lrow[8];
#pragma unroll
  for (int r = 0; r < 8; ++r) { mrow[r] = -3.402823466e38f; lrow[r] = 0.f; }

  for (int j0 = 0; j0 <= q0; j0 += 128) {
    // Stage K block [key][hd] (async) and V block transposed [hd][key] (sync).
#pragma unroll
    for (int c = tid; c < 2048; c += 256) {
      int r = c >> 4, off = (c & 15) * 8;
      cp16_async(&Ks[r][off],
                 &Kf[(size_t)(b * S + j0 + r) * D + hoff + off]);
      bf16x8 v = *(const bf16x8*)&Vf[(size_t)(b * S + j0 + r) * D + hoff + off];
#pragma unroll
      for (int e = 0; e < 8; ++e) Vt[off + e][r] = v[e];
    }
    async_wait0();
    __syncthreads();

    // Scores: 16 q-rows x 128 keys in 8 WMMA column-chunks.
    f32x8 sc[8];
#pragma unroll
    for (int nc = 0; nc < 8; ++nc) {
      f32x8 a = zero8();
#pragma unroll
      for (int kc = 0; kc < 4; ++kc) {
        Frag bf_;
        const __bf16* bp = &Ks[nc * 16 + l15][kc * 32 + lhalf * 8];
        bf_.h[0] = *(const bf16x8*)bp;
        bf_.h[1] = *(const bf16x8*)(bp + 16);
        a = wmma_bf16(qf[kc].v, bf_.v, a);
      }
      sc[nc] = a;
    }

    if (j0 == q0) {  // diagonal block: mask key > query
#pragma unroll
      for (int nc = 0; nc < 8; ++nc) {
        int key = nc * 16 + l15;
#pragma unroll
        for (int r = 0; r < 8; ++r) {
          int qr = wave * 16 + r + lhalf * 8;
          if (key > qr) sc[nc][r] = -3.402823466e38f;
        }
      }
    }

    // Online softmax. Row stats per lane-half (rows r + lhalf*8), reduce
    // across the 16 lanes of the half with xor-shuffles 1,2,4,8.
#pragma unroll
    for (int r = 0; r < 8; ++r) {
      float m = sc[0][r];
#pragma unroll
      for (int nc = 1; nc < 8; ++nc) m = fmaxf(m, sc[nc][r]);
      m = fmaxf(m, __shfl_xor(m, 1, 32));
      m = fmaxf(m, __shfl_xor(m, 2, 32));
      m = fmaxf(m, __shfl_xor(m, 4, 32));
      m = fmaxf(m, __shfl_xor(m, 8, 32));
      float mnew = fmaxf(mrow[r], m);
      float s = 0.f;
#pragma unroll
      for (int nc = 0; nc < 8; ++nc) {
        float e = __expf(sc[nc][r] - mnew);
        sc[nc][r] = e;
        s += e;
      }
      s += __shfl_xor(s, 1, 32);
      s += __shfl_xor(s, 2, 32);
      s += __shfl_xor(s, 4, 32);
      s += __shfl_xor(s, 8, 32);
      float alpha = __expf(mrow[r] - mnew);
      lrow[r] = lrow[r] * alpha + s;
      mrow[r] = mnew;
#pragma unroll
      for (int nh = 0; nh < 8; ++nh) ctx[nh][r] *= alpha;
    }

    // P: C-layout -> LDS -> A-frag layout (per-wave slice, in-order LDS).
#pragma unroll
    for (int nc = 0; nc < 8; ++nc)
#pragma unroll
      for (int r = 0; r < 8; ++r)
        Ps[wave][r + lhalf * 8][nc * 16 + l15] = (__bf16)sc[nc][r];

    // ctx += P(16x128) @ V(128x128)
#pragma unroll
    for (int kc = 0; kc < 4; ++kc) {
      Frag pf;
      const __bf16* pp = &Ps[wave][l15][kc * 32 + lhalf * 8];
      pf.h[0] = *(const bf16x8*)pp;
      pf.h[1] = *(const bf16x8*)(pp + 16);
#pragma unroll
      for (int nh = 0; nh < 8; ++nh) {
        Frag vf_;
        const __bf16* vp = &Vt[nh * 16 + l15][kc * 32 + lhalf * 8];
        vf_.h[0] = *(const bf16x8*)vp;
        vf_.h[1] = *(const bf16x8*)(vp + 16);
        ctx[nh] = wmma_bf16(pf.v, vf_.v, ctx[nh]);
      }
    }
    __syncthreads();
  }

  // Normalize and write merged-head ctx.
#pragma unroll
  for (int r = 0; r < 8; ++r) {
    float inv = 1.0f / lrow[r];
    size_t row = (size_t)(b * S + q0 + wave * 16 + r + lhalf * 8) * D + hoff;
#pragma unroll
    for (int nh = 0; nh < 8; ++nh)
      O[row + nh * 16 + l15] = (__bf16)(ctx[nh][r] * inv);
  }
}

// ---------------------------------------------------------------------------
extern "C" void kernel_launch(void* const* d_in, const int* in_sizes, int n_in,
                              void* d_out, int out_size, void* d_ws, size_t ws_size,
                              hipStream_t stream) {
  const float* X  = (const float*)d_in[0];
  const float* Wq = (const float*)d_in[1];
  const float* Wk = (const float*)d_in[2];
  const float* Wv = (const float*)d_in[3];
  const float* Wo = (const float*)d_in[4];
  const float* bo = (const float*)d_in[5];

  const int Bb = 2, S = 2048, D = 2048, H = 16;
  const int M = Bb * S;  // 4096

  // Workspace (bf16): X | Wqt Wkt Wvt Wot | Q K V ; ctx reuses X's slot.
  __bf16* Xb  = (__bf16*)d_ws;
  __bf16* Wqt = Xb  + (size_t)M * D;
  __bf16* Wkt = Wqt + (size_t)D * D;
  __bf16* Wvt = Wkt + (size_t)D * D;
  __bf16* Wot = Wvt + (size_t)D * D;
  __bf16* Qb  = Wot + (size_t)D * D;
  __bf16* Kb  = Qb  + (size_t)M * D;
  __bf16* Vb  = Kb  + (size_t)M * D;
  __bf16* Ctx = Xb;  // X dead after QKV projections

  cvt_f32_bf16<<<(M * D) / (4 * 256), 256, 0, stream>>>(X, Xb, M * D);

  dim3 tgrid(D / 64, D / 64);
  transpose_cvt<<<tgrid, 256, 0, stream>>>(Wq, Wqt, D, D);
  transpose_cvt<<<tgrid, 256, 0, stream>>>(Wk, Wkt, D, D);
  transpose_cvt<<<tgrid, 256, 0, stream>>>(Wv, Wvt, D, D);
  transpose_cvt<<<tgrid, 256, 0, stream>>>(Wo, Wot, D, D);

  dim3 ggrid(D / 128, M / 128);
  gemm_bf16_wmma<<<ggrid, 256, 0, stream>>>(Xb, Wqt, Qb, nullptr, nullptr, M, D, D);
  gemm_bf16_wmma<<<ggrid, 256, 0, stream>>>(Xb, Wkt, Kb, nullptr, nullptr, M, D, D);
  gemm_bf16_wmma<<<ggrid, 256, 0, stream>>>(Xb, Wvt, Vb, nullptr, nullptr, M, D, D);

  dim3 agrid(S / 128, H, Bb);
  attn_flash_wmma<<<agrid, 256, 0, stream>>>(Qb, Kb, Vb, Ctx, S, D, H);

  gemm_bf16_wmma<<<ggrid, 256, 0, stream>>>(Ctx, Wot, nullptr, (float*)d_out, bo,
                                            M, D, D);
}
